// NodeDetector_77979426226959
// MI455X (gfx1250) — matrix operand order repaired and checked by reading, not compiled, not run on hardware
//
#include <hip/hip_runtime.h>
#include <math.h>

typedef _Float16 f16;
typedef _Float16 v16h __attribute__((ext_vector_type(16)));
typedef float    v8f  __attribute__((ext_vector_type(8)));

#define N_NODES 256
#define C2      64     // GAT hidden dim
#define HC      128    // NUM_HEAD * C2
#define ODIM    64

// ---------------------------------------------------------------------------
// Precompute 1: base = E_emb@emb_proj@conv_w0 + (x@node_proj)@conv_w1 + conv_b
// Stores base[256,128] and x_w1[256,128].
// ---------------------------------------------------------------------------
__global__ __launch_bounds__(128)
void k_prep1(const float* __restrict__ x, const float* __restrict__ Eemb,
             const float* __restrict__ node_proj, const float* __restrict__ emb_proj,
             const float* __restrict__ conv_w0, const float* __restrict__ conv_w1,
             const float* __restrict__ conv_b,
             float* __restrict__ baseg, float* __restrict__ xw1g) {
    __shared__ float sx[64], sE[64], sxp[128], sEp[128];
    const int r = blockIdx.x, c = threadIdx.x;
    if (c < 64) { sx[c] = x[r * 64 + c]; sE[c] = Eemb[r * 64 + c]; }
    __syncthreads();
    float xp = 0.f, ep = 0.f;
    for (int k = 0; k < 64; ++k) {
        xp += sx[k] * node_proj[k * 128 + c];
        ep += sE[k] * emb_proj[k * 128 + c];
    }
    sxp[c] = xp; sEp[c] = ep;
    __syncthreads();
    float w1 = 0.f, b0 = 0.f;
    for (int k = 0; k < 128; ++k) {
        w1 += sxp[k] * conv_w1[k * 128 + c];
        b0 += sEp[k] * conv_w0[k * 128 + c];
    }
    xw1g[r * 128 + c]  = w1;
    baseg[r * 128 + c] = b0 + w1 + conv_b[c];
}

// ---------------------------------------------------------------------------
// Precompute 2 (row-wise): P0[r] = (tanh(base[r])@lin2 + b)@normal_proj
//                          PM[r] = (tanh(base[r]-xw1[r])@lin2 + b)@masked_proj
// ---------------------------------------------------------------------------
__global__ __launch_bounds__(128)
void k_prep2(const float* __restrict__ baseg, const float* __restrict__ xw1g,
             const float* __restrict__ lin2_w, const float* __restrict__ lin2_b,
             const float* __restrict__ masked_proj, const float* __restrict__ normal_proj,
             float* __restrict__ P0, float* __restrict__ PM) {
    __shared__ float h0[128], hm[128], m0[64], mm[64];
    const int r = blockIdx.x, c = threadIdx.x;
    float b = baseg[r * 128 + c], w = xw1g[r * 128 + c];
    h0[c] = tanhf(b);
    hm[c] = tanhf(b - w);
    __syncthreads();
    if (c < 64) {
        float a0 = lin2_b[c], am = lin2_b[c];
        for (int k = 0; k < 128; ++k) {
            a0 += h0[k] * lin2_w[k * 64 + c];
            am += hm[k] * lin2_w[k * 64 + c];
        }
        m0[c] = a0; mm[c] = am;
    }
    __syncthreads();
    if (c < 64) {
        float p0 = 0.f, pm = 0.f;
        for (int k = 0; k < 64; ++k) {
            p0 += m0[k] * normal_proj[k * 64 + c];
            pm += mm[k] * masked_proj[k * 64 + c];
        }
        P0[r * 64 + c] = p0;
        PM[r * 64 + c] = pm;
    }
}

// ---------------------------------------------------------------------------
// One-time deterministic CSR-by-dst build (single thread; E ~ 4352).
// ---------------------------------------------------------------------------
__global__ void k_csr(const int* __restrict__ dst, int E,
                      int* __restrict__ csrs, int* __restrict__ csre) {
    if (threadIdx.x == 0 && blockIdx.x == 0) {
        int cnt[N_NODES];
        for (int n = 0; n < N_NODES; ++n) cnt[n] = 0;
        for (int e = 0; e < E; ++e) cnt[dst[e]]++;
        int run = 0;
        for (int n = 0; n < N_NODES; ++n) {
            csrs[n] = run; run += cnt[n]; cnt[n] = csrs[n];
        }
        csrs[N_NODES] = run;
        for (int e = 0; e < E; ++e) csre[cnt[dst[e]]++] = e;
    }
}

// ---------------------------------------------------------------------------
// Main kernel: one workgroup (256 thr = 8 wave32) per masked node.
// Per-block global scratch (floats): xlh[f16 16384f] xrh[f16 16384f]
//                                    bufB[16384f] logit[2E f]
// GEMMs via v_wmma_f32_16x16x32_f16, A from LDS (f16), B = transposed weight
// panel in LDS (f16), f32 accumulators, bias fused at store.
// ---------------------------------------------------------------------------
__global__ __launch_bounds__(256)
void k_gat(const int* __restrict__ src, const int* __restrict__ dst,
           const int* __restrict__ csrs, const int* __restrict__ csre,
           const float* __restrict__ P0, const float* __restrict__ PM,
           const float* __restrict__ wl1, const float* __restrict__ bl1,
           const float* __restrict__ wr1, const float* __restrict__ br1,
           const float* __restrict__ att1, const float* __restrict__ gb1,
           const float* __restrict__ wl2, const float* __restrict__ bl2,
           const float* __restrict__ wr2, const float* __restrict__ br2,
           const float* __restrict__ att2, const float* __restrict__ gb2,
           const float* __restrict__ rec_w, const float* __restrict__ rec_b,
           float* __restrict__ out, float* __restrict__ blkbase,
           long strideF, int E, int numNodes) {
    __shared__ f16 sProj[N_NODES * C2];   // 32 KB: current layer input, f16
    __shared__ f16 sW[HC * C2];           // 16 KB: transposed weight [n][k]

    const int tid  = threadIdx.x;
    const int lane = tid & 31;
    const int wave = tid >> 5;
    const int hi   = (lane >> 4) & 1;
    const int l15  = lane & 15;

    float* blk   = blkbase + (long)blockIdx.x * strideF;
    f16*   xlh   = (f16*)blk;             // 256*128 halves (16384 floats)
    f16*   xrh   = (f16*)(blk + 16384);   // 256*128 halves
    float* bufB  = blk + 32768;           // 256*64 f32
    float* logit = blk + 49152;           // 2*E f32

    const float* WLp[2] = {wl1, wl2};
    const float* BLp[2] = {bl1, bl2};
    const float* WRp[2] = {wr1, wr2};
    const float* BRp[2] = {br1, br2};
    const float* ATp[2] = {att1, att2};
    const float* GBp[2] = {gb1, gb2};

    for (int node = blockIdx.x; node < numNodes; node += gridDim.x) {
        __syncthreads();
        for (int layer = 0; layer < 2; ++layer) {
            // ---- stage layer input (f16) into LDS; layer0 swaps in masked row
            if (layer == 0) {
                for (int idx = tid; idx < N_NODES * C2; idx += blockDim.x) {
                    int m = idx >> 6;
                    sProj[idx] = (f16)((m == node) ? PM[idx] : P0[idx]);
                }
            } else {
                for (int idx = tid; idx < N_NODES * C2; idx += blockDim.x)
                    sProj[idx] = (f16)bufB[idx];
            }
            __syncthreads();

            // ---- xl = proj@wl + bl ; xr = proj@wr + br  (WMMA)
            for (int side = 0; side < 2; ++side) {
                const float* W = side ? WRp[layer] : WLp[layer];
                const float* B = side ? BRp[layer] : BLp[layer];
                f16* dmat = side ? xrh : xlh;
                // transpose-stage weight panel: sW[n*64+k] = W[k*128+n]
                for (int idx = tid; idx < HC * C2; idx += blockDim.x) {
                    int n = idx & 127, k = idx >> 7;
                    sW[n * C2 + k] = (f16)W[k * HC + n];
                }
                __syncthreads();
                for (int t = wave; t < 128; t += 8) {   // 16x16 tiles: 16(M)x8(N)
                    int mt = t >> 3, nt = t & 7;
                    v8f acc = {0.f, 0.f, 0.f, 0.f, 0.f, 0.f, 0.f, 0.f};
                    for (int k0 = 0; k0 < 64; k0 += 32) {
                        v16h a, b;
                        int arow = (mt * 16 + l15) * C2;
#pragma unroll
                        for (int v = 0; v < 8; ++v) {   // A 16-bit 16x32 layout
                            int ka = k0 + ((2 * v < 8) ? 2 * v : 2 * v + 8) + hi * 8;
                            a[2 * v]     = sProj[arow + ka];
                            a[2 * v + 1] = sProj[arow + ka + 1];
                        }
                        int brow = (nt * 16 + l15) * C2;
#pragma unroll
                        for (int v = 0; v < 8; ++v) {   // B 16-bit 32x16 layout
                            int kb = k0 + ((2 * v < 8) ? 2 * v : 2 * v + 8) + hi * 8;
                            b[2 * v]     = sW[brow + kb];
                            b[2 * v + 1] = sW[brow + kb + 1];
                        }
                        acc = __builtin_amdgcn_wmma_f32_16x16x32_f16(
                            false, a, false, b, (short)0, acc, false, false);
                    }
                    int ncol = nt * 16 + l15;
                    float bv = B[ncol];
#pragma unroll
                    for (int r = 0; r < 8; ++r) {       // C/D f32 16x16 layout
                        int mrow = mt * 16 + hi * 8 + r;
                        dmat[mrow * HC + ncol] = (f16)(acc[r] + bv);
                    }
                }
                __syncthreads();
            }

            // ---- per-edge logits: att . leaky_relu(xl[src] + xr[dst])
            const float* ATT = ATp[layer];
            for (int e = tid; e < E; e += blockDim.x) {
                int s = src[e], d = dst[e];
                const f16* xs = xlh + s * HC;
                const f16* xd = xrh + d * HC;
                float a0 = 0.f, a1 = 0.f;
                for (int c = 0; c < C2; ++c) {
                    float v0 = (float)xs[c]      + (float)xd[c];
                    float v1 = (float)xs[64 + c] + (float)xd[64 + c];
                    v0 = v0 > 0.f ? v0 : 0.2f * v0;
                    v1 = v1 > 0.f ? v1 : 0.2f * v1;
                    a0 += ATT[c]      * v0;
                    a1 += ATT[64 + c] * v1;
                }
                logit[e * 2]     = a0;
                logit[e * 2 + 1] = a1;
            }
            __syncthreads();

            // ---- per-target softmax + weighted aggregation (thread t = node t)
            {
                int n = tid;
                int j0 = csrs[n], j1 = csrs[n + 1];
                float mx0 = -1e30f, mx1 = -1e30f;
                for (int j = j0; j < j1; ++j) {
                    int e = csre[j];
                    mx0 = fmaxf(mx0, logit[e * 2]);
                    mx1 = fmaxf(mx1, logit[e * 2 + 1]);
                }
                float d0 = 0.f, d1 = 0.f;
                for (int j = j0; j < j1; ++j) {
                    int e = csre[j];
                    d0 += __expf(logit[e * 2]     - mx0);
                    d1 += __expf(logit[e * 2 + 1] - mx1);
                }
                float i0 = 1.f / (d0 + 1e-16f), i1 = 1.f / (d1 + 1e-16f);
                const float* GB = GBp[layer];
                for (int cc = 0; cc < C2; cc += 16) {
                    float a0[16], a1[16];
#pragma unroll
                    for (int q = 0; q < 16; ++q) { a0[q] = 0.f; a1[q] = 0.f; }
                    for (int j = j0; j < j1; ++j) {
                        int e = csre[j];
                        int s = src[e];
                        float w0 = __expf(logit[e * 2]     - mx0) * i0;
                        float w1 = __expf(logit[e * 2 + 1] - mx1) * i1;
                        const f16* xs = xlh + s * HC + cc;
#pragma unroll
                        for (int q = 0; q < 16; ++q) {
                            a0[q] += w0 * (float)xs[q];
                            a1[q] += w1 * (float)xs[64 + q];
                        }
                    }
#pragma unroll
                    for (int q = 0; q < 16; ++q) {
                        float v = 0.5f * (a0[q] + a1[q]) + GB[cc + q];
                        v = v > 0.f ? v : (__expf(v) - 1.f);   // elu
                        bufB[n * C2 + cc + q] = v;
                    }
                }
            }
            __syncthreads();
        } // layers

        // ---- reconstruction: out[node] = tanh(g2[node] @ rec_w + rec_b)
        if (tid < ODIM) {
            const float* g = bufB + node * C2;
            float acc = rec_b[tid];
            for (int k = 0; k < C2; ++k) acc += g[k] * rec_w[k * ODIM + tid];
            out[node * ODIM + tid] = tanhf(acc);
        }
        __syncthreads();
    }
}

// ---------------------------------------------------------------------------
extern "C" void kernel_launch(void* const* d_in, const int* in_sizes, int n_in,
                              void* d_out, int out_size, void* d_ws, size_t ws_size,
                              hipStream_t stream) {
    const float* x           = (const float*)d_in[0];
    const float* Eemb        = (const float*)d_in[1];
    const int*   eidx        = (const int*)d_in[2];
    const float* node_proj   = (const float*)d_in[3];
    const float* emb_proj    = (const float*)d_in[4];
    const float* conv_w0     = (const float*)d_in[5];
    const float* conv_w1     = (const float*)d_in[6];
    const float* conv_b      = (const float*)d_in[7];
    const float* lin2_w      = (const float*)d_in[8];
    const float* lin2_b      = (const float*)d_in[9];
    const float* masked_proj = (const float*)d_in[10];
    const float* normal_proj = (const float*)d_in[11];
    const float* g1_wl = (const float*)d_in[12];
    const float* g1_bl = (const float*)d_in[13];
    const float* g1_wr = (const float*)d_in[14];
    const float* g1_br = (const float*)d_in[15];
    const float* g1_att  = (const float*)d_in[16];
    const float* g1_bias = (const float*)d_in[17];
    const float* g2_wl = (const float*)d_in[18];
    const float* g2_bl = (const float*)d_in[19];
    const float* g2_wr = (const float*)d_in[20];
    const float* g2_br = (const float*)d_in[21];
    const float* g2_att  = (const float*)d_in[22];
    const float* g2_bias = (const float*)d_in[23];
    const float* rec_w = (const float*)d_in[24];
    const float* rec_b = (const float*)d_in[25];

    const int E = in_sizes[2] / 2;
    const int* srcp = eidx;
    const int* dstp = eidx + E;

    float* ws    = (float*)d_ws;
    float* baseg = ws;                 // 256*128
    float* xw1g  = ws + 32768;         // 256*128
    float* P0    = ws + 65536;         // 256*64
    float* PM    = ws + 81920;         // 256*64
    int*   csrs  = (int*)(ws + 98304); // 257 ints (padded to 512)
    int*   csre  = (int*)(ws + 98816); // E ints
    long headerF = 98816 + (long)((E + 127) & ~127);
    long strideF = (49152L + 2L * E + 255L) & ~255L; // per-block scratch floats
    float* blkbase = ws + headerF;

    size_t headerBytes = (size_t)headerF * 4;
    size_t strideBytes = (size_t)strideF * 4;
    int nblk = 1;
    if (ws_size > headerBytes + strideBytes) {
        long m = (long)((ws_size - headerBytes) / strideBytes);
        nblk = (int)(m < 256 ? m : 256);
    }
    if (nblk < 1) nblk = 1;

    k_prep1<<<N_NODES, 128, 0, stream>>>(x, Eemb, node_proj, emb_proj,
                                         conv_w0, conv_w1, conv_b, baseg, xw1g);
    k_prep2<<<N_NODES, 128, 0, stream>>>(baseg, xw1g, lin2_w, lin2_b,
                                         masked_proj, normal_proj, P0, PM);
    k_csr<<<1, 32, 0, stream>>>(dstp, E, csrs, csre);
    k_gat<<<nblk, 256, 0, stream>>>(srcp, dstp, csrs, csre, P0, PM,
                                    g1_wl, g1_bl, g1_wr, g1_br, g1_att, g1_bias,
                                    g2_wl, g2_bl, g2_wr, g2_br, g2_att, g2_bias,
                                    rec_w, rec_b, (float*)d_out, blkbase,
                                    strideF, E, N_NODES);
}
